// VectorQuantizer_24584392802479
// MI455X (gfx1250) — matrix operand order repaired and checked by reading, not compiled, not run on hardware
//
#include <hip/hip_runtime.h>
#include <hip/hip_bf16.h>

typedef __bf16 bf16;
typedef __attribute__((ext_vector_type(16))) __bf16 v16bf;
typedef __attribute__((ext_vector_type(8)))  __bf16 v8bf;
typedef __attribute__((ext_vector_type(8)))  float  v8f;

#define N_CODES       1024
#define EMB           256
#define ROWS_PER_BLK  128
#define CODES_PER_BLK 128
#define LDS_STRIDE    264      // bf16 elems: 528 B row pitch, 16B aligned
#define SPATIAL       32768    // 32*32*32
#define CH_STRIDE     32768
#define BATCH_STRIDE  8388608  // 256*32768
#define NUMEL         33554432 // 4*256*32768
#define BETA          0.25f

// Monotone float -> uint key (total order, works for +/-, inf)
__device__ __forceinline__ unsigned fkey(float f) {
    unsigned u = __float_as_uint(f);
    return (u & 0x80000000u) ? ~u : (u | 0x80000000u);
}

// -------- prep: fp32 code norms + bf16 codebook ----------------------------
__global__ __launch_bounds__(256) void vq_prep(const float* __restrict__ cb,
                                               float* __restrict__ cnorm,
                                               bf16* __restrict__ cbf) {
    int j = blockIdx.x * 256 + threadIdx.x;
    if (j < N_CODES) {
        float s = 0.0f;
        #pragma unroll 8
        for (int k = 0; k < EMB; ++k) {
            float v = cb[j * EMB + k];
            s += v * v;
            cbf[j * EMB + k] = (bf16)v;
        }
        cnorm[j] = s;
    }
}

// -------- main: bf16 WMMA distance GEMM + argmin + gather + loss -----------
__global__ __launch_bounds__(256) void vq_argmin(const float* __restrict__ x,
                                                 const float* __restrict__ cbf32,
                                                 const float* __restrict__ cnorm,
                                                 const uint4* __restrict__ cbf_u4,
                                                 float* __restrict__ out,
                                                 float* __restrict__ partials) {
    __shared__ bf16 Abf[ROWS_PER_BLK * LDS_STRIDE];
    __shared__ bf16 Bbf[CODES_PER_BLK * LDS_STRIDE];
    __shared__ float cnorm_s[N_CODES];
    __shared__ unsigned long long rowbest[ROWS_PER_BLK];
    __shared__ float red[256];

    const int tid  = threadIdx.x;
    const int lane = tid & 31;
    const int l16  = lane & 15;
    const int lhi  = lane >> 4;            // 0 or 1 (lane half)
    const int wv   = tid >> 5;             // wave 0..7
    const int wm   = (wv >> 1) * 32;       // wave row base: 0,32,64,96
    const int wn   = (wv & 1)  * 64;       // wave col base: 0 or 64

    const int r0 = blockIdx.x * ROWS_PER_BLK;   // global row base
    const int bb = r0 >> 15;                    // batch index
    const int s0 = r0 & (SPATIAL - 1);          // spatial base

    for (int j = tid; j < N_CODES; j += 256)      cnorm_s[j] = cnorm[j];
    for (int j = tid; j < ROWS_PER_BLK; j += 256) rowbest[j] = ~0ULL;

    // ---- Load A tile: x[(bb, k, s0+sl)] -> bf16 LDS [sl][k] ----
    {
        const int sl = tid & 127;
        const int kh = tid >> 7;                 // k parity (0/1)
        const float* xb = x + (size_t)bb * BATCH_STRIDE + s0 + sl;
        #pragma unroll 4
        for (int kk = 0; kk < EMB; kk += 2) {
            int k = kk + kh;
            Abf[sl * LDS_STRIDE + k] = (bf16)xb[(size_t)k * CH_STRIDE];
        }
    }
    __syncthreads();

    float    minval[16];
    unsigned minidx[16];
    #pragma unroll
    for (int i = 0; i < 16; ++i) { minval[i] = __builtin_inff(); minidx[i] = 0u; }

    const v8f vzero = {0.f,0.f,0.f,0.f,0.f,0.f,0.f,0.f};

    for (int cblk = 0; cblk < N_CODES / CODES_PER_BLK; ++cblk) {
        // ---- Load B tile: bf16 codebook rows [cblk*128, +128) -> LDS ----
        for (int c = tid; c < CODES_PER_BLK * (EMB / 8); c += 256) {
            int code = c >> 5;                   // 32 x 16B chunks per code
            int kc   = c & 31;
            *(uint4*)&Bbf[code * LDS_STRIDE + kc * 8] =
                cbf_u4[(size_t)(cblk * CODES_PER_BLK + code) * 32 + kc];
        }
        __syncthreads();

        v8f acc[2][4];
        #pragma unroll
        for (int t = 0; t < 2; ++t)
            #pragma unroll
            for (int c = 0; c < 4; ++c) acc[t][c] = vzero;

        #pragma unroll
        for (int kb = 0; kb < EMB; kb += 32) {
            // A fragments (ISA 16-bit A 16x32 layout): lanes 0-15 K=+0..7/+16..23,
            // lanes 16-31 K=+8..15/+24..31, M = l16
            v16bf afr[2];
            #pragma unroll
            for (int t = 0; t < 2; ++t) {
                const bf16* p = &Abf[(wm + t * 16 + l16) * LDS_STRIDE + kb + lhi * 8];
                v8bf lo = *(const v8bf*)p;
                v8bf hi = *(const v8bf*)(p + 16);
                #pragma unroll
                for (int i = 0; i < 8; ++i) { afr[t][i] = lo[i]; afr[t][i + 8] = hi[i]; }
            }
            // B fragments (32x16): lane N = l16, K = kb + lhi*8 + 0..7 / +16..23
            v16bf bfr[4];
            #pragma unroll
            for (int c = 0; c < 4; ++c) {
                const bf16* p = &Bbf[(wn + c * 16 + l16) * LDS_STRIDE + kb + lhi * 8];
                v8bf lo = *(const v8bf*)p;
                v8bf hi = *(const v8bf*)(p + 16);
                #pragma unroll
                for (int i = 0; i < 8; ++i) { bfr[c][i] = lo[i]; bfr[c][i + 8] = hi[i]; }
            }
            #pragma unroll
            for (int t = 0; t < 2; ++t)
                #pragma unroll
                for (int c = 0; c < 4; ++c)
                    acc[t][c] = __builtin_amdgcn_wmma_f32_16x16x32_bf16(
                        false, afr[t], false, bfr[c], (short)0, acc[t][c], false, false);
        }

        // ---- fold dist = ||c||^2 - 2*dot into running per-thread min ----
        #pragma unroll
        for (int c = 0; c < 4; ++c) {
            int   coln = cblk * CODES_PER_BLK + wn + c * 16 + l16;
            float cn2  = cnorm_s[coln];
            #pragma unroll
            for (int t = 0; t < 2; ++t)
                #pragma unroll
                for (int r = 0; r < 8; ++r) {
                    float dist = cn2 - 2.0f * acc[t][c][r];
                    int   slot = t * 8 + r;
                    bool  lt   = dist < minval[slot];   // strict: first index wins
                    minval[slot] = lt ? dist : minval[slot];
                    minidx[slot] = lt ? (unsigned)coln : minidx[slot];
                }
        }
        __syncthreads();   // B tile reuse guard
    }

    // ---- merge per-thread mins: packed (key,idx) u64 LDS atomic min ----
    #pragma unroll
    for (int t = 0; t < 2; ++t)
        #pragma unroll
        for (int r = 0; r < 8; ++r) {
            int slot = t * 8 + r;
            int row  = wm + t * 16 + lhi * 8 + r;   // C-tile lane->row mapping
            unsigned long long pk =
                ((unsigned long long)fkey(minval[slot]) << 32) |
                (unsigned long long)minidx[slot];
            atomicMin(&rowbest[row], pk);
        }
    __syncthreads();

    // ---- gather quant, write output (x layout), accumulate loss ----
    float lsum = 0.0f;
    {
        const int sl = tid & 127;
        const int kh = tid >> 7;
        const size_t base = (size_t)bb * BATCH_STRIDE + s0 + sl;
        unsigned idx = (unsigned)(rowbest[sl] & 0xFFFFFFFFull);
        const float* crow = cbf32 + (size_t)idx * EMB;
        #pragma unroll 4
        for (int kk = 0; kk < EMB; kk += 2) {
            int k = kk + kh;
            size_t g = base + (size_t)k * CH_STRIDE;
            float q  = crow[k];
            float xv = x[g];
            out[g] = q;
            float d = q - xv;
            lsum += d * d;
        }
    }
    red[tid] = lsum;
    __syncthreads();
    #pragma unroll
    for (int s = 128; s > 0; s >>= 1) {
        if (tid < s) red[tid] += red[tid + s];
        __syncthreads();
    }
    if (tid == 0)
        partials[blockIdx.x] = red[0] * ((1.0f + BETA) / (float)NUMEL);
}

// -------- deterministic final loss reduction -------------------------------
__global__ __launch_bounds__(256) void vq_loss(const float* __restrict__ partials,
                                               float* __restrict__ out_loss) {
    __shared__ float red[256];
    float s = 0.0f;
    for (int i = threadIdx.x; i < 1024; i += 256) s += partials[i];  // fixed order
    red[threadIdx.x] = s;
    __syncthreads();
    #pragma unroll
    for (int st = 128; st > 0; st >>= 1) {
        if (threadIdx.x < st) red[threadIdx.x] += red[threadIdx.x + st];
        __syncthreads();
    }
    if (threadIdx.x == 0) *out_loss = red[0];
}

extern "C" void kernel_launch(void* const* d_in, const int* in_sizes, int n_in,
                              void* d_out, int out_size, void* d_ws, size_t ws_size,
                              hipStream_t stream) {
    const float* x  = (const float*)d_in[0];   // (4,256,32,32,32) f32
    const float* cb = (const float*)d_in[1];   // (1024,256) f32
    float* out = (float*)d_out;                // quant (NUMEL) + loss (1)

    // ws layout: [0,4K) cnorm f32 | [4K, 4K+512K) bf16 codebook | [+4K) partials
    float* cnorm    = (float*)d_ws;
    bf16*  cbf      = (bf16*)((char*)d_ws + 4096);
    float* partials = (float*)((char*)d_ws + 4096 + (size_t)N_CODES * EMB * 2);

    vq_prep<<<(N_CODES + 255) / 256, 256, 0, stream>>>(cb, cnorm, cbf);
    vq_argmin<<<131072 / ROWS_PER_BLK, 256, 0, stream>>>(
        x, cb, cnorm, (const uint4*)cbf, out, partials);
    vq_loss<<<1, 256, 0, stream>>>(partials, out + NUMEL);
}
